// RouterSinkhorn_32418413150245
// MI455X (gfx1250) — matrix operand order repaired and checked by reading, not compiled, not run on hardware
//
#include <hip/hip_runtime.h>
#include <hip/hip_bf16.h>

// ---------------------------------------------------------------------------
// RouterSinkhorn for MI455X (gfx1250, wave32, WMMA)
//   logits = hidden(8192x4096) @ W(64x4096)^T  via split-bf16 WMMA
//     (a = a_hi + a_lo, b = b_hi + b_lo; keep hi*hi + hi*lo + lo*hi)
//   - W pre-split to bf16 planes in global (L2-resident); GEMM stages them to
//     LDS with GLOBAL_LOAD_ASYNC_TO_LDS_B128 (ASYNCcnt, no VGPR round-trip)
//   - hidden split to bf16 planes during LDS staging with native v_cvt bf16
//   - compute path: raw ds_load_b128 fragments -> v_wmma_f32_16x16x32_bf16
// ---------------------------------------------------------------------------

#define T_TOK 8192
#define H_DIM 4096
#define E_EXP 64
#define KC    32            // K per LDS stage = one bf16 WMMA K-step
#define MBLK  64            // tokens per block
#define KSTR  40            // LDS row stride in bf16 (32 + 8 pad, 80 B, 16B-mult)
#define NKB   (H_DIM / KC)
#define SINKHORN_ITERS 30
#define SINK_EPS 1e-8f

typedef __attribute__((ext_vector_type(16))) __bf16 v16bf;
typedef __attribute__((ext_vector_type(8)))  float  v8f;

union FragBF {               // 16 bf16 fragment = two 16-byte LDS loads
    uint4 q[2];
    v16bf v;
};
union BFPack8 {              // 8 bf16 <-> uint4 (one b128 store)
    __bf16 b[8];
    uint4  q;
};
union BFPack4 {              // 4 bf16 <-> uint2
    __bf16 b[4];
    uint2  q;
};

// Split 8 fp32 into bf16 hi plane + bf16 lo plane (native cvt, RNE)
__device__ __forceinline__ void split8(const float4 a, const float4 b,
                                       uint4& hi, uint4& lo) {
    float f[8] = {a.x, a.y, a.z, a.w, b.x, b.y, b.z, b.w};
    BFPack8 H, L;
    #pragma unroll
    for (int i = 0; i < 8; ++i) {
        __bf16 h = (__bf16)f[i];
        H.b[i] = h;
        L.b[i] = (__bf16)(f[i] - (float)h);
    }
    hi = H.q;
    lo = L.q;
}

// gfx1250 async global->LDS copy (ASYNCcnt); lds_off = LDS byte address VGPR
__device__ __forceinline__ void async_copy_b128(unsigned lds_off,
                                                const void* gptr) {
    asm volatile("global_load_async_to_lds_b128 %0, %1, off"
                 :: "v"(lds_off), "v"(gptr)
                 : "memory");
}
__device__ __forceinline__ void wait_async0() {
    asm volatile("s_wait_asynccnt 0" ::: "memory");
}

// ---------------------------------------------------------------------------
// Pre-kernel: split W (64x4096 fp32) into bf16 hi/lo planes (row-major).
// 65536 threads, one float4 each.
// ---------------------------------------------------------------------------
__global__ void wsplit_kernel(const float* __restrict__ W,
                              unsigned short* __restrict__ whi,
                              unsigned short* __restrict__ wlo) {
    const size_t i = (size_t)blockIdx.x * blockDim.x + threadIdx.x; // < 65536
    const float4 v = *(const float4*)&W[i * 4];
    float f[4] = {v.x, v.y, v.z, v.w};
    BFPack4 H, L;
    #pragma unroll
    for (int j = 0; j < 4; ++j) {
        __bf16 h = (__bf16)f[j];
        H.b[j] = h;
        L.b[j] = (__bf16)(f[j] - (float)h);
    }
    *(uint2*)&whi[i * 4] = H.q;
    *(uint2*)&wlo[i * 4] = L.q;
}

// ---------------------------------------------------------------------------
// GEMM: 128 blocks x 256 threads (8 waves). Block tile: 64 tokens x 64 experts.
// Wave w: M-tile = w>>1, N-tiles {(w&1)*2, (w&1)*2+1}.
// ---------------------------------------------------------------------------
__launch_bounds__(256)
__global__ void router_gemm_kernel(const float* __restrict__ hidden,
                                   const unsigned short* __restrict__ whi,
                                   const unsigned short* __restrict__ wlo,
                                   float* __restrict__ logits,
                                   float* __restrict__ aff,
                                   float* __restrict__ cexp) {
    // bf16 planes in LDS; row stride 40 bf16 = 80 B (16B multiple, bank-skewed)
    __shared__ __align__(16) unsigned short Ahi[2][MBLK][KSTR];
    __shared__ __align__(16) unsigned short Alo[2][MBLK][KSTR];
    __shared__ __align__(16) unsigned short Bhi[2][E_EXP][KSTR];
    __shared__ __align__(16) unsigned short Blo[2][E_EXP][KSTR];

    const int tid   = threadIdx.x;
    const int wave  = tid >> 5;
    const int lane  = tid & 31;
    const int row16 = lane & 15;
    const int half  = lane >> 4;
    const int tBase = blockIdx.x * MBLK;

    const int mtile  = wave >> 1;
    const int ntile0 = (wave & 1) * 2;

    v8f acc[2];
    acc[0] = (v8f)0.0f;
    acc[1] = (v8f)0.0f;

    // thread -> (row, k-octet) map for staging: 64 rows x 4 octets of 8
    const int srow = tid >> 2;
    const int sk8  = (tid & 3) * 8;

    auto stage_chunk = [&](int kb, int b) {
        const int k0 = kb * KC;
        // B planes: pure bf16 copy via async global->LDS (no VGPR round-trip)
        {
            const size_t off = (size_t)srow * H_DIM + k0 + sk8;
            async_copy_b128((unsigned)(size_t)&Bhi[b][srow][sk8], &whi[off]);
            async_copy_b128((unsigned)(size_t)&Blo[b][srow][sk8], &wlo[off]);
        }
        // A planes: load 8 fp32, native-cvt split, store hi/lo b128
        {
            const float* src = &hidden[(size_t)(tBase + srow) * H_DIM + k0 + sk8];
            float4 a0 = *(const float4*)src;
            float4 a1 = *(const float4*)(src + 4);
            uint4 hi, lo;
            split8(a0, a1, hi, lo);
            *(uint4*)&Ahi[b][srow][sk8] = hi;
            *(uint4*)&Alo[b][srow][sk8] = lo;
        }
    };

    stage_chunk(0, 0);
    wait_async0();
    __syncthreads();

    for (int kb = 0; kb < NKB; ++kb) {
        const int b = kb & 1;
        if (kb + 1 < NKB) stage_chunk(kb + 1, b ^ 1);
        if (kb + 2 < NKB) {
            // gfx1250 prefetch path (global_prefetch_b8)
            const int kn = (kb + 2) * KC;
            __builtin_prefetch(&hidden[(size_t)(tBase + srow) * H_DIM + kn + sk8], 0, 0);
        }

        // ---- A fragments: lane holds row M=lane%16,
        //      elements 0..7 = K=half*8+{0..7}, 8..15 = K=16+half*8+{0..7}
        FragBF ahi, alo;
        {
            const int rA = mtile * 16 + row16;
            ahi.q[0] = *(const uint4*)&Ahi[b][rA][half * 8];
            ahi.q[1] = *(const uint4*)&Ahi[b][rA][16 + half * 8];
            alo.q[0] = *(const uint4*)&Alo[b][rA][half * 8];
            alo.q[1] = *(const uint4*)&Alo[b][rA][16 + half * 8];
        }

        #pragma unroll
        for (int nt = 0; nt < 2; ++nt) {
            // ---- B fragments: lane holds col N=lane%16, K=half*8+{0..7}, 16+half*8+{0..7}
            const int ew = (ntile0 + nt) * 16 + row16;
            FragBF bhi, blo;
            bhi.q[0] = *(const uint4*)&Bhi[b][ew][half * 8];
            bhi.q[1] = *(const uint4*)&Bhi[b][ew][16 + half * 8];
            blo.q[0] = *(const uint4*)&Blo[b][ew][half * 8];
            blo.q[1] = *(const uint4*)&Blo[b][ew][16 + half * 8];

            acc[nt] = __builtin_amdgcn_wmma_f32_16x16x32_bf16(
                false, ahi.v, false, bhi.v, (short)0, acc[nt], false, false);
            acc[nt] = __builtin_amdgcn_wmma_f32_16x16x32_bf16(
                false, ahi.v, false, blo.v, (short)0, acc[nt], false, false);
            acc[nt] = __builtin_amdgcn_wmma_f32_16x16x32_bf16(
                false, alo.v, false, bhi.v, (short)0, acc[nt], false, false);
        }

        // publish chunk kb+1: wait for our async B copies, then barrier
        wait_async0();
        __syncthreads();
    }

    // ---- Epilogue: D layout => VGPR r: token = half*8 + r, expert = lane%16
    #pragma unroll
    for (int nt = 0; nt < 2; ++nt) {
        const int e = (ntile0 + nt) * 16 + row16;
        #pragma unroll
        for (int r = 0; r < 8; ++r) {
            const int tok = tBase + mtile * 16 + half * 8 + r;
            const float v = acc[nt][r];
            const size_t o = (size_t)tok * E_EXP + e;
            logits[o] = v;
            aff[o]    = 1.0f / (1.0f + expf(-v));
            cexp[o]   = expf(v);
        }
    }
}

// ---------------------------------------------------------------------------
// Sinkhorn
// ---------------------------------------------------------------------------
__global__ void sinkhorn_init(float* __restrict__ d0, float* __restrict__ d1) {
    int i = blockIdx.x * blockDim.x + threadIdx.x;
    if (i < T_TOK) d0[i] = 1.0f;
    if (i < E_EXP) d1[i] = 1.0f;
}

// d0[t] = 1 / (T * (sum_e d1[e]*c[t,e] + eps))
__global__ void sinkhorn_row(const float* __restrict__ c,
                             const float* __restrict__ d1,
                             float* __restrict__ d0) {
    int t = blockIdx.x * blockDim.x + threadIdx.x;
    if (t >= T_TOK) return;
    const float4* cr = (const float4*)(c + (size_t)t * E_EXP);
    const float4* dv = (const float4*)d1;
    float s = 0.0f;
    #pragma unroll
    for (int i = 0; i < E_EXP / 4; ++i) {
        float4 cv = cr[i];
        float4 d  = dv[i];
        s += cv.x * d.x + cv.y * d.y + cv.z * d.z + cv.w * d.w;
    }
    d0[t] = 1.0f / ((float)T_TOK * (s + SINK_EPS));
}

// d1[e] = 1 / (E * (sum_t d0[t]*c[t,e] + eps)); one block per expert
__global__ void sinkhorn_col(const float* __restrict__ c,
                             const float* __restrict__ d0,
                             float* __restrict__ d1) {
    __shared__ float red[256];
    const int e = blockIdx.x;
    float s = 0.0f;
    for (int t = threadIdx.x; t < T_TOK; t += 256)
        s += d0[t] * c[(size_t)t * E_EXP + e];
    red[threadIdx.x] = s;
    __syncthreads();
    for (int w = 128; w > 0; w >>= 1) {
        if (threadIdx.x < w) red[threadIdx.x] += red[threadIdx.x + w];
        __syncthreads();
    }
    if (threadIdx.x == 0)
        d1[e] = 1.0f / ((float)E_EXP * (red[0] + SINK_EPS));
}

// expert_index[t] = argmax_e d1[e]*c[t,e]*d0[t]  (first max wins, like jnp)
__global__ void sinkhorn_argmax(const float* __restrict__ c,
                                const float* __restrict__ d0,
                                const float* __restrict__ d1,
                                long long* __restrict__ idx) {
    int t = blockIdx.x * blockDim.x + threadIdx.x;
    if (t >= T_TOK) return;
    const float d0t = d0[t];
    float best = -3.4e38f;
    int be = 0;
    #pragma unroll
    for (int e = 0; e < E_EXP; ++e) {
        float v = d1[e] * c[(size_t)t * E_EXP + e] * d0t;
        if (v > best) { best = v; be = e; }
    }
    idx[t] = (long long)be;
}

// ---------------------------------------------------------------------------
extern "C" void kernel_launch(void* const* d_in, const int* in_sizes, int n_in,
                              void* d_out, int out_size, void* d_ws, size_t ws_size,
                              hipStream_t stream) {
    (void)in_sizes; (void)n_in; (void)out_size; (void)ws_size;

    const float* hidden = (const float*)d_in[0];   // (8192, 4096) f32
    const float* W      = (const float*)d_in[1];   // (64, 4096)   f32

    float* out    = (float*)d_out;
    float* logits = out;                                   // (T,E) f32
    float* affs   = out + (size_t)T_TOK * E_EXP;           // (T,E) f32
    long long* idx = (long long*)(out + 2 * (size_t)T_TOK * E_EXP); // (T,1) i64

    float* ws = (float*)d_ws;
    float* c  = ws;                                  // (T,E) f32 : exp(logits)
    float* d0 = ws + (size_t)T_TOK * E_EXP;          // (T,)
    float* d1 = d0 + T_TOK;                          // (E,)
    unsigned short* whi = (unsigned short*)(d1 + E_EXP);       // (E,H) bf16 hi
    unsigned short* wlo = whi + (size_t)E_EXP * H_DIM;         // (E,H) bf16 lo

    wsplit_kernel<<<(E_EXP * H_DIM / 4) / 256, 256, 0, stream>>>(W, whi, wlo);
    router_gemm_kernel<<<T_TOK / MBLK, 256, 0, stream>>>(hidden, whi, wlo,
                                                         logits, affs, c);
    sinkhorn_init<<<(T_TOK + 255) / 256, 256, 0, stream>>>(d0, d1);
    for (int it = 0; it < SINKHORN_ITERS; ++it) {
        sinkhorn_row<<<T_TOK / 256, 256, 0, stream>>>(c, d1, d0);
        sinkhorn_col<<<E_EXP, 256, 0, stream>>>(c, d0, d1);
    }
    sinkhorn_argmax<<<T_TOK / 256, 256, 0, stream>>>(c, d0, d1, idx);
}
